// Stage2_3DRefine_69449621176961
// MI455X (gfx1250) — compile-verified, matrix operand order checked by reading, not run on hardware
//
#include <hip/hip_runtime.h>
#include <hip/hip_bf16.h>

typedef __attribute__((ext_vector_type(16))) _Float16 v16h;
typedef __attribute__((ext_vector_type(8)))  _Float16 v8h;
typedef __attribute__((ext_vector_type(8)))  float    v8f;
typedef __attribute__((ext_vector_type(4))) unsigned int u32x4;
typedef __attribute__((ext_vector_type(8))) int          i32x8;
typedef __attribute__((ext_vector_type(4))) int          i32x4;

// Problem dims (from reference setup_inputs)
constexpr int B_  = 2, V_ = 4, X_ = 64, Y_ = 64, Z_ = 32;
constexpr int C_  = 16, H_ = 37, W_ = 37;
constexpr int N_  = X_ * Y_ * Z_;     // 131072 voxels per batch
constexpr long BN = (long)B_ * N_;    // 262144 rows
constexpr int D_  = 384;              // hidden dim
constexpr int KPAD = 32;              // K=17 padded to one 16x16x32 WMMA step
constexpr int NT  = D_ / 16;          // 24 N-tiles of W1
constexpr int W1H_ELTS = NT * 32 * 16;          // 12288 f16 = 24 KB

// ---------------------------------------------------------------------------
// Kernel 0: rearrange W1 (17x384 f32, row-major) into f16 WMMA B-fragments.
// Fragment layout per tile t: lane l holds column n = t*16 + (l&15);
// lanes 0-15 hold K = {0..7, 16..23}, lanes 16-31 hold K = {8..15, 24..31}.
// Stored contiguously: w1h[((t*32 + l)*16 + e)]  (16 f16 per lane per tile).
// ---------------------------------------------------------------------------
__global__ void prep_w1_kernel(const float* __restrict__ W1,
                               _Float16* __restrict__ w1h) {
    int idx = blockIdx.x * blockDim.x + threadIdx.x;
    if (idx >= W1H_ELTS) return;
    int e = idx & 15;
    int l = (idx >> 4) & 31;
    int t = idx >> 9;
    int half = l >> 4;
    int k = (e < 8 ? e : e + 8) + half * 8;   // K index within padded 32
    int n = t * 16 + (l & 15);
    float v = (k < 17) ? W1[k * D_ + n] : 0.0f;
    w1h[idx] = (_Float16)v;
}

// ---------------------------------------------------------------------------
// Kernel 1: multi-view bilinear sampling -> padded f16 activation rows.
// One thread per voxel. Feature maps are ~700KB total => L2-resident gathers.
// Row layout (32 f16): [iv, feat/(16*nvalid) x16, zeros x15]
// ---------------------------------------------------------------------------
__global__ void __launch_bounds__(256)
sample_kernel(const float* __restrict__ iv,
              const float* __restrict__ dino,
              const float* __restrict__ proj,
              const float* __restrict__ valid,
              _Float16* __restrict__ xin) {
    long m = (long)blockIdx.x * blockDim.x + threadIdx.x;
    if (m >= BN) return;
    int b = (int)(m / N_);
    int n = (int)(m % N_);

    float acc[C_];
#pragma unroll
    for (int c = 0; c < C_; ++c) acc[c] = 0.0f;
    float nv = 0.0f;

    for (int v = 0; v < V_; ++v) {
        long vi = (long)(b * V_ + v) * N_ + n;
        nv += valid[vi];
        float px = proj[vi * 2 + 0];
        float py = proj[vi * 2 + 1];
        // grid_sample, align_corners=False, padding_mode='zeros'
        float fx = (px + 1.0f) * (W_ * 0.5f) - 0.5f;
        float fy = (py + 1.0f) * (H_ * 0.5f) - 0.5f;
        float x0 = floorf(fx), y0 = floorf(fy);
        float wx1 = fx - x0, wy1 = fy - y0;
        float wx0 = 1.0f - wx1, wy0 = 1.0f - wy1;
        int ix0 = (int)x0, iy0 = (int)y0;
        int ix1 = ix0 + 1, iy1 = iy0 + 1;
        bool vx0 = (ix0 >= 0) & (ix0 <= W_ - 1);
        bool vx1 = (ix1 >= 0) & (ix1 <= W_ - 1);
        bool vy0 = (iy0 >= 0) & (iy0 <= H_ - 1);
        bool vy1 = (iy1 >= 0) & (iy1 <= H_ - 1);
        int cx0 = min(max(ix0, 0), W_ - 1), cx1 = min(max(ix1, 0), W_ - 1);
        int cy0 = min(max(iy0, 0), H_ - 1), cy1 = min(max(iy1, 0), H_ - 1);
        float w00 = (vx0 && vy0) ? wx0 * wy0 : 0.0f;
        float w10 = (vx1 && vy0) ? wx1 * wy0 : 0.0f;
        float w01 = (vx0 && vy1) ? wx0 * wy1 : 0.0f;
        float w11 = (vx1 && vy1) ? wx1 * wy1 : 0.0f;
        const float* img = dino + (long)(b * V_ + v) * C_ * H_ * W_;
        int o00 = cy0 * W_ + cx0, o10 = cy0 * W_ + cx1;
        int o01 = cy1 * W_ + cx0, o11 = cy1 * W_ + cx1;
#pragma unroll
        for (int c = 0; c < C_; ++c) {
            const float* ic = img + c * (H_ * W_);
            acc[c] += ic[o00] * w00 + ic[o10] * w10 +
                      ic[o01] * w01 + ic[o11] * w11;
        }
    }

    nv = fmaxf(nv, 1.0f);
    float scale = (1.0f / 16.0f) / nv;   // /nvalid then /DOWNSAMPLE_DINO_DIM

    _Float16 row[KPAD];
    row[0] = (_Float16)iv[m];
#pragma unroll
    for (int c = 0; c < C_; ++c) row[1 + c] = (_Float16)(acc[c] * scale);
#pragma unroll
    for (int k = 17; k < KPAD; ++k) row[k] = (_Float16)0.0f;

    v8h r0, r1, r2, r3;
#pragma unroll
    for (int i = 0; i < 8; ++i) {
        r0[i] = row[i];      r1[i] = row[8 + i];
        r2[i] = row[16 + i]; r3[i] = row[24 + i];
    }
    v8h* dst = (v8h*)(xin + m * KPAD);
    dst[0] = r0; dst[1] = r1; dst[2] = r2; dst[3] = r3;
}

// ---------------------------------------------------------------------------
// Branch-free tanh-GELU using hardware transcendentals:
//   gelu(x) = 0.5*x*(1+tanh(g)) = x - x * rcp(exp2(2.302118752*inner) + 1)
// v_exp_f32 / v_rcp_f32 are TRANS32 and co-execute with the WMMA pipe.
// ---------------------------------------------------------------------------
__device__ __forceinline__ float gelu_fast(float x) {
    float x2 = x * x;
    float inner = __builtin_fmaf(0.044715f, x2 * x, x);    // x + 0.044715 x^3
    // 0.7978845608 * 2 * log2(e) = 2.302118751862
    float t = __builtin_amdgcn_exp2f(2.302118751862f * inner);
    float r = __builtin_amdgcn_rcpf(t + 1.0f);
    return x - x * r;
}

// ---------------------------------------------------------------------------
// Kernel 2: fused WMMA MLP. 8 waves/block; each wave owns a 16-row M-tile.
// Block prologue: wave 0 issues one TDM (tensor_load_to_lds) pulling the
// whole 24KB W1 fragment buffer into LDS (TENSORcnt), then barrier.
// Per N-tile: ds_load B-fragment, bias-preloaded f32 accumulator, one
// v_wmma_f32_16x16x32_f16, GELU, FMA with W2 slice; butterfly-reduce over N.
// ---------------------------------------------------------------------------
__global__ void __launch_bounds__(256)
mlp_kernel(const _Float16* __restrict__ xin,
           const _Float16* __restrict__ w1h,
           const float* __restrict__ b1,
           const float* __restrict__ W2,
           const float* __restrict__ b2,
           const float* __restrict__ iv,
           const unsigned char* __restrict__ smask,
           float* __restrict__ out) {
    __shared__ _Float16 w1s[W1H_ELTS];   // 24 KB of B-fragments

    int wave = threadIdx.x >> 5;
    int lane = threadIdx.x & 31;
    int lm   = lane & 15;
    int half = lane >> 4;

    // ---- TDM: DMA w1h (global) -> w1s (LDS), one descriptor per block ----
    if (threadIdx.x < 32) {
        unsigned long long ga = (unsigned long long)(uintptr_t)w1h;
        // Flat LDS aperture: addr[31:0] is the LDS byte offset.
        unsigned int lds_off = (unsigned int)(uintptr_t)(&w1s[0]);
        // D# group 0: count=1 | lds_addr | global_addr[56:0] | type=2
        u32x4 g0;
        g0[0] = 1u;                                   // count=1, no gather
        g0[1] = lds_off;                              // lds_addr
        g0[2] = (unsigned int)(ga & 0xFFFFFFFFull);   // global_addr[31:0]
        g0[3] = (unsigned int)((ga >> 32) & 0x1FFFFFFull) | 0x80000000u; // [56:32] | type=2
        // D# group 1: 1-D transfer of 12288 f16 elements
        i32x8 g1;
        g1[0] = (1 << 16);                    // workgroup_mask=0, data_size=1 (2B)
        g1[1] = (int)((W1H_ELTS & 0xFFFF) << 16);          // tensor_dim0[15:0]
        g1[2] = (int)((W1H_ELTS >> 16) | (1 << 16));       // tensor_dim0[31:16] | tensor_dim1=1
        g1[3] = (int)((W1H_ELTS & 0xFFFF) << 16);          // tensor_dim1 hi=0 | tile_dim0
        g1[4] = 1;                                         // tile_dim1=1, tile_dim2=0
        g1[5] = (int)W1H_ELTS;                             // tensor_dim0_stride[31:0]
        g1[6] = 0;                                         // stride0[47:32] | stride1 lo
        g1[7] = 0;                                         // stride1 hi
        i32x4 g2 = {0, 0, 0, 0};                           // <=2-D: groups 2/3 unused
        i32x4 g3 = {0, 0, 0, 0};
        i32x8 g4 = {0, 0, 0, 0, 0, 0, 0, 0};               // clang-23 6-arg form
        __builtin_amdgcn_tensor_load_to_lds(g0, g1, g2, g3, g4, 0);
        __builtin_amdgcn_s_wait_tensorcnt(0);
    }
    __syncthreads();

    long tile = (long)blockIdx.x * 8 + wave;
    long m0 = tile * 16;

    const _Float16* rowp = xin + (m0 + lm) * KPAD + half * 8;
    v8h alo = *(const v8h*)(rowp);        // K 0..7   or 8..15
    v8h ahi = *(const v8h*)(rowp + 16);   // K 16..23 or 24..31
    v16h A = __builtin_shufflevector(alo, ahi,
                                     0, 1, 2, 3, 4, 5, 6, 7,
                                     8, 9, 10, 11, 12, 13, 14, 15);
    float acc[8];
#pragma unroll
    for (int j = 0; j < 8; ++j) acc[j] = 0.0f;

#pragma unroll 4
    for (int t = 0; t < NT; ++t) {
        const _Float16* bp = &w1s[(t * 32 + lane) << 4];   // LDS fragments
        v8h blo = *(const v8h*)(bp);
        v8h bhi = *(const v8h*)(bp + 8);
        v16h Bm = __builtin_shufflevector(blo, bhi,
                                          0, 1, 2, 3, 4, 5, 6, 7,
                                          8, 9, 10, 11, 12, 13, 14, 15);
        float bias = b1[t * 16 + lm];
        v8f cc;
#pragma unroll
        for (int j = 0; j < 8; ++j) cc[j] = bias;
        // D = A*B + C ; emits v_wmma_f32_16x16x32_f16
        cc = __builtin_amdgcn_wmma_f32_16x16x32_f16(
                false, A, false, Bm, (short)0, cc, false, false);
        float w2v = W2[t * 16 + lm];
#pragma unroll
        for (int j = 0; j < 8; ++j) acc[j] += gelu_fast(cc[j]) * w2v;
    }

    // Reduce over the 16 N-lanes within each half (wave32 butterfly)
#pragma unroll
    for (int s = 1; s < 16; s <<= 1) {
#pragma unroll
        for (int j = 0; j < 8; ++j) acc[j] += __shfl_xor(acc[j], s, 32);
    }

    // lane 0 -> rows 0..7, lane 16 -> rows 8..15 (VGPR j = M j + 8*half)
    if (lm == 0) {
        float bb = b2[0];
        long rbase = m0 + (long)half * 8;
#pragma unroll
        for (int j = 0; j < 8; ++j) {
            long mm = rbase + j;
            float o = acc[j] + bb + iv[mm];     // bias + residual
            o = (o >= 0.0f) ? o : 0.01f * o;    // leaky_relu
            o = smask[mm] ? o : 0.0f;           // sparse scatter mask
            out[mm] = o;
        }
    }
}

// ---------------------------------------------------------------------------
// Launch: d_in order = {input_volume, dino_features, projection_indices,
//                       valid_projection_indices, sparse_mask, W1, b1, W2, b2}
// d_ws: [0, BN*32 f16)        padded activation rows (16 MB, L2-resident)
//       [BN*32 f16, +12288)   W1 B-fragments (24 KB)
// ---------------------------------------------------------------------------
extern "C" void kernel_launch(void* const* d_in, const int* in_sizes, int n_in,
                              void* d_out, int out_size, void* d_ws, size_t ws_size,
                              hipStream_t stream) {
    const float* iv    = (const float*)d_in[0];
    const float* dino  = (const float*)d_in[1];
    const float* proj  = (const float*)d_in[2];
    const float* valid = (const float*)d_in[3];
    const unsigned char* smask = (const unsigned char*)d_in[4];
    const float* W1 = (const float*)d_in[5];
    const float* b1 = (const float*)d_in[6];
    const float* W2 = (const float*)d_in[7];
    const float* b2 = (const float*)d_in[8];
    float* out = (float*)d_out;

    _Float16* xin = (_Float16*)d_ws;
    _Float16* w1h = xin + (size_t)BN * KPAD;

    prep_w1_kernel<<<(W1H_ELTS + 255) / 256, 256, 0, stream>>>(W1, w1h);
    sample_kernel<<<(int)(BN / 256), 256, 0, stream>>>(iv, dino, proj, valid, xin);
    // BN/16 = 16384 M-tiles, 8 waves per block -> 2048 blocks
    mlp_kernel<<<(int)(BN / 16 / 8), 256, 0, stream>>>(xin, w1h, b1, W2, b2,
                                                       iv, smask, out);
}
